// HandwritingRNN_89352499626020
// MI455X (gfx1250) — matrix operand order repaired and checked by reading, not compiled, not run on hardware
//
#include <hip/hip_runtime.h>

// ---------------------------------------------------------------- constants
#define NBLK 25
#define NTHR 256
#define NWAVE (NBLK * NTHR / 32)   // 200 wave32 waves

#define BB   64     // batch
#define TT   600    // timesteps
#define TCC  50     // char seq length
#define ALPH 60     // alphabet
#define HID  400    // lstm hidden
#define H4   1600   // 4*HID
#define KATT 10
#define NOUT 121

// unified activation layout (columns of X, all 16-aligned):
//   [0,3)   x_t          (pad to 16)
//   [16,76) window       (pad to 80)
//   [80,480)  h1
//   [480,880) h2
//   [896,1296) h3        (pad to 1312)
#define LDX 1312
#define OFF_WIN 16
#define OFF_H1  80
#define OFF_H2  480
#define OFF_H3  896
#define K1 480
#define K2 896
#define K3 1312
#define KB1 15   // 480/32
#define KB2 28   // 896/32
#define KB3 41   // 1312/32

// ---------------------------------------------------------------- workspace
constexpr size_t aln(size_t x) { return (x + 255) & ~size_t(255); }
constexpr size_t B_BAR = 0;                                   // 2 uints
constexpr size_t B_X   = aln(B_BAR + 64);
constexpr size_t B_W1  = aln(B_X  + (size_t)BB * LDX * 2);    // bf16
constexpr size_t B_W2  = aln(B_W1 + (size_t)H4 * K1 * 2);
constexpr size_t B_W3  = aln(B_W2 + (size_t)H4 * K2 * 2);
constexpr size_t B_WG  = aln(B_W3 + (size_t)H4 * K3 * 2);
constexpr size_t B_G   = aln(B_WG + (size_t)128 * K3 * 2);    // f32 from here
constexpr size_t B_C1  = aln(B_G  + (size_t)BB * H4 * 4);
constexpr size_t B_C2  = aln(B_C1 + (size_t)BB * HID * 4);
constexpr size_t B_C3  = aln(B_C2 + (size_t)BB * HID * 4);
constexpr size_t B_Z   = aln(B_C3 + (size_t)BB * HID * 4);
constexpr size_t B_KAP = aln(B_Z  + (size_t)BB * 128 * 4);
constexpr size_t B_P   = aln(B_KAP + (size_t)BB * KATT * 4);
constexpr size_t B_PHI = aln(B_P  + (size_t)BB * 30 * 4);
constexpr size_t B_B1  = aln(B_PHI + (size_t)BB * TCC * 4);
constexpr size_t B_B2  = aln(B_B1 + H4 * 4);
constexpr size_t B_B3  = aln(B_B2 + H4 * 4);
constexpr size_t B_BG  = aln(B_B3 + H4 * 4);
constexpr size_t B_END = aln(B_BG + NOUT * 4);

// ---------------------------------------------------------------- helpers
typedef __attribute__((ext_vector_type(16))) __bf16 v16bf;
typedef __attribute__((ext_vector_type(8)))  float  v8f;

__device__ inline unsigned short f2bf(float f) {
  unsigned u = __float_as_uint(f);
  unsigned r = u + 0x7FFFu + ((u >> 16) & 1u);   // RNE
  return (unsigned short)(r >> 16);
}
__device__ inline float bf2f(unsigned short h) {
  return __uint_as_float(((unsigned)h) << 16);
}
__device__ inline float sigm(float x) { return 1.f / (1.f + expf(-x)); }

struct alignas(16) U4x { unsigned x, y, z, w; };
struct U8x { U4x a, b; };
__device__ inline v16bf ldfrag(const unsigned short* p0, const unsigned short* p1) {
  U8x r;
  r.a = *(const U4x*)p0;
  r.b = *(const U4x*)p1;
  return __builtin_bit_cast(v16bf, r);
}
__device__ inline v8f wmma_bf16(v16bf a, v16bf b, v8f c) {
  return __builtin_amdgcn_wmma_f32_16x16x32_bf16(false, a, false, b, (short)0, c,
                                                 false, false);
}

// sense-reversing grid barrier (persistent kernel, all blocks resident)
__device__ inline void gbar(unsigned* bar) {
  __syncthreads();
  if (threadIdx.x == 0) {
    __threadfence();
    unsigned gen = __hip_atomic_load(&bar[1], __ATOMIC_RELAXED, __HIP_MEMORY_SCOPE_AGENT);
    unsigned prev = __hip_atomic_fetch_add(&bar[0], 1u, __ATOMIC_ACQ_REL, __HIP_MEMORY_SCOPE_AGENT);
    if (prev == NBLK - 1) {
      __hip_atomic_store(&bar[0], 0u, __ATOMIC_RELAXED, __HIP_MEMORY_SCOPE_AGENT);
      __hip_atomic_store(&bar[1], gen + 1u, __ATOMIC_RELEASE, __HIP_MEMORY_SCOPE_AGENT);
    } else {
      while (__hip_atomic_load(&bar[1], __ATOMIC_ACQUIRE, __HIP_MEMORY_SCOPE_AGENT) == gen)
        __builtin_amdgcn_s_sleep(2);
    }
  }
  __syncthreads();
}

// ---------------------------------------------------------------- GEMM phase
// Gout[64 x LDN] = X[64 x (KB*32)] @ W^T ;  W stored as [N rows x LDK] bf16.
// Unit = (n-strip of 16, m-half of 32 rows) -> 2 accumulators, B reused.
template <int KB, int LDK, int LDN>
__device__ void mm_phase(const unsigned short* __restrict__ X,
                         const unsigned short* __restrict__ W,
                         float* __restrict__ Gout, int Ntiles, int gwave, int lane) {
  const int ln = lane & 15, hf = lane >> 4;
  for (int u = gwave; u < Ntiles * 2; u += NWAVE) {
    const int s = u >> 1, mh = u & 1;
    const unsigned short* a0 = X + (size_t)(mh * 32 + ln) * LDX;
    const unsigned short* a1 = a0 + (size_t)16 * LDX;
    const unsigned short* bp = W + (size_t)(s * 16 + ln) * LDK + hf * 8;
    __builtin_prefetch(bp, 0, 1);
    v8f acc0 = {0.f, 0.f, 0.f, 0.f, 0.f, 0.f, 0.f, 0.f};
    v8f acc1 = {0.f, 0.f, 0.f, 0.f, 0.f, 0.f, 0.f, 0.f};
    for (int kb = 0; kb < KB; ++kb) {
      const unsigned short* ak0 = a0 + kb * 32 + hf * 8;   // A 16x32 bf16 layout
      const unsigned short* ak1 = a1 + kb * 32 + hf * 8;
      v16bf fb  = ldfrag(bp + kb * 32, bp + kb * 32 + 16);  // B 32x16 bf16 layout
      v16bf fa0 = ldfrag(ak0, ak0 + 16);
      v16bf fa1 = ldfrag(ak1, ak1 + 16);
      acc0 = wmma_bf16(fa0, fb, acc0);
      acc1 = wmma_bf16(fa1, fb, acc1);
    }
    float* g0 = Gout + (size_t)(mh * 32 + hf * 8) * LDN + s * 16 + ln;
    float* g1 = g0 + (size_t)16 * LDN;
#pragma unroll
    for (int r = 0; r < 8; ++r) {
      g0[(size_t)r * LDN] = acc0[r];
      g1[(size_t)r * LDN] = acc1[r];
    }
  }
}

// ---------------------------------------------------------------- elementwise
__device__ inline void lstm_ew(const float* __restrict__ G, const float* __restrict__ bias,
                               float* __restrict__ C, unsigned short* __restrict__ Xh,
                               int tid, int stride) {
  for (int i = tid; i < BB * HID; i += stride) {
    int b = i / HID, j = i - b * HID;
    const float* gr = G + (size_t)b * H4;
    float gi = gr[j] + bias[j];
    float gf = gr[HID + j] + bias[HID + j];
    float gg = gr[2 * HID + j] + bias[2 * HID + j];
    float go = gr[3 * HID + j] + bias[3 * HID + j];
    float c = sigm(gf) * C[i] + sigm(gi) * tanhf(gg);
    C[i] = c;
    Xh[(size_t)b * LDX + j] = f2bf(sigm(go) * tanhf(c));
  }
}

__device__ inline void out_ew(int t, const float* __restrict__ Z,
                              const float* __restrict__ BG, float* __restrict__ out,
                              int tid, int stride) {
  for (int b = tid; b < BB; b += stride) {
    const float* zr = Z + (size_t)b * 128;
    float mx = -1e30f;
    for (int j = 0; j < 20; ++j) mx = fmaxf(mx, zr[1 + j] + BG[1 + j]);
    float s = 0.f;
    for (int j = 0; j < 20; ++j) s += expf(zr[1 + j] + BG[1 + j] - mx);
    float inv = 1.f / s;
    float* o = out + ((size_t)b * TT + t) * NOUT;
    for (int j = 0; j < 20; ++j) o[j] = expf(zr[1 + j] + BG[1 + j] - mx) * inv;
    for (int j = 0; j < 40; ++j) o[20 + j] = zr[21 + j] + BG[21 + j];
    for (int j = 0; j < 40; ++j) o[60 + j] = expf(zr[61 + j] + BG[61 + j]);
    for (int j = 0; j < 20; ++j) o[100 + j] = tanhf(zr[101 + j] + BG[101 + j]);
    o[120] = sigm(zr[0] + BG[0]);
  }
}

// ---------------------------------------------------------------- prep kernels
__global__ void k_zero(unsigned* ws32) {
  size_t n1 = B_W1 / 4;                  // barrier + X region
  size_t n2 = (B_END - B_G) / 4;         // all f32 state
  size_t gid = (size_t)blockIdx.x * blockDim.x + threadIdx.x;
  size_t gsz = (size_t)gridDim.x * blockDim.x;
  for (size_t i = gid; i < n1 + n2; i += gsz) {
    size_t w = (i < n1) ? i : (B_G / 4 + (i - n1));
    ws32[w] = 0u;
  }
}

__global__ void k_bias(const float* b1a, const float* b1b, const float* b2a,
                       const float* b2b, const float* b3a, const float* b3b,
                       const float* bg, unsigned char* ws) {
  int i = blockIdx.x * blockDim.x + threadIdx.x;
  if (i < H4) {
    ((float*)(ws + B_B1))[i] = b1a[i] + b1b[i];
    ((float*)(ws + B_B2))[i] = b2a[i] + b2b[i];
    ((float*)(ws + B_B3))[i] = b3a[i] + b3b[i];
  }
  if (i < NOUT) ((float*)(ws + B_BG))[i] = bg[i];
}

__global__ void k_w1(const float* Wih, const float* Whh, unsigned char* ws) {
  int i = blockIdx.x * blockDim.x + threadIdx.x;
  if (i >= H4 * K1) return;
  int n = i / K1, k = i - n * K1;
  float v = 0.f;
  if (k < 3) v = Wih[n * 63 + 60 + k];                         // x
  else if (k >= OFF_WIN && k < OFF_WIN + 60) v = Wih[n * 63 + (k - OFF_WIN)];  // window
  else if (k >= OFF_H1 && k < OFF_H1 + HID) v = Whh[n * HID + (k - OFF_H1)];   // h1
  ((unsigned short*)(ws + B_W1))[i] = f2bf(v);
}

__global__ void k_w2(const float* Wih, const float* Whh, unsigned char* ws) {
  int i = blockIdx.x * blockDim.x + threadIdx.x;
  if (i >= H4 * K2) return;
  int n = i / K2, k = i - n * K2;
  float v = 0.f;
  if (k < 3) v = Wih[n * 463 + k];                                              // x
  else if (k >= OFF_WIN && k < OFF_WIN + 60) v = Wih[n * 463 + 403 + (k - OFF_WIN)];
  else if (k >= OFF_H1 && k < OFF_H1 + HID) v = Wih[n * 463 + 3 + (k - OFF_H1)];
  else if (k >= OFF_H2 && k < OFF_H2 + HID) v = Whh[n * HID + (k - OFF_H2)];
  ((unsigned short*)(ws + B_W2))[i] = f2bf(v);
}

__global__ void k_w3(const float* Wih, const float* Whh, unsigned char* ws) {
  int i = blockIdx.x * blockDim.x + threadIdx.x;
  if (i >= H4 * K3) return;
  int n = i / K3, k = i - n * K3;
  float v = 0.f;
  if (k < 3) v = Wih[n * 863 + k];
  else if (k >= OFF_WIN && k < OFF_WIN + 60) v = Wih[n * 863 + 803 + (k - OFF_WIN)];
  else if (k >= OFF_H1 && k < OFF_H1 + HID) v = Wih[n * 863 + 3 + (k - OFF_H1)];
  else if (k >= OFF_H2 && k < OFF_H2 + HID) v = Wih[n * 863 + 403 + (k - OFF_H2)];
  else if (k >= OFF_H3 && k < OFF_H3 + HID) v = Whh[n * HID + (k - OFF_H3)];
  ((unsigned short*)(ws + B_W3))[i] = f2bf(v);
}

__global__ void k_wg(const float* Wg, unsigned char* ws) {
  int i = blockIdx.x * blockDim.x + threadIdx.x;
  if (i >= 128 * K3) return;
  int n = i / K3, k = i - n * K3;
  float v = 0.f;
  if (n < NOUT) {
    if (k >= OFF_H1 && k < OFF_H1 + HID) v = Wg[n * 1200 + (k - OFF_H1)];
    else if (k >= OFF_H2 && k < OFF_H2 + HID) v = Wg[n * 1200 + 400 + (k - OFF_H2)];
    else if (k >= OFF_H3 && k < OFF_H3 + HID) v = Wg[n * 1200 + 800 + (k - OFF_H3)];
  }
  ((unsigned short*)(ws + B_WG))[i] = f2bf(v);
}

// ---------------------------------------------------------------- persistent RNN
__global__ __launch_bounds__(NTHR) void k_rnn(const float* __restrict__ xin,
                                              const int* __restrict__ cseq,
                                              const int* __restrict__ clen,
                                              const float* __restrict__ Watt,
                                              const float* __restrict__ batt,
                                              unsigned char* __restrict__ ws,
                                              float* __restrict__ out) {
  unsigned* bar = (unsigned*)(ws + B_BAR);
  unsigned short* X = (unsigned short*)(ws + B_X);
  const unsigned short* W1 = (const unsigned short*)(ws + B_W1);
  const unsigned short* W2 = (const unsigned short*)(ws + B_W2);
  const unsigned short* W3 = (const unsigned short*)(ws + B_W3);
  const unsigned short* WG = (const unsigned short*)(ws + B_WG);
  float* G   = (float*)(ws + B_G);
  float* C1  = (float*)(ws + B_C1);
  float* C2  = (float*)(ws + B_C2);
  float* C3  = (float*)(ws + B_C3);
  float* Z   = (float*)(ws + B_Z);
  float* KAP = (float*)(ws + B_KAP);
  float* P   = (float*)(ws + B_P);
  float* PHI = (float*)(ws + B_PHI);
  const float* B1 = (const float*)(ws + B_B1);
  const float* B2 = (const float*)(ws + B_B2);
  const float* B3 = (const float*)(ws + B_B3);
  const float* BG = (const float*)(ws + B_BG);

  const int tid   = threadIdx.x;
  const int lane  = tid & 31;
  const int gwave = blockIdx.x * (NTHR / 32) + (tid >> 5);
  const int gtid  = blockIdx.x * NTHR + tid;

  for (int t = 0; t < TT; ++t) {
    // ---- Phase A: write x_t into X; emit output for step t-1 (block 0)
    if (blockIdx.x == 0) {
      if (t > 0) out_ew(t - 1, Z, BG, out, tid, NTHR);
      for (int i = tid; i < BB * 3; i += NTHR) {
        int b = i / 3, c = i - b * 3;
        X[(size_t)b * LDX + c] = f2bf(xin[((size_t)b * TT + t) * 3 + c]);
      }
    }
    gbar(bar);

    // ---- Phase B: LSTM1 gates   g = X[:, :480] @ W1c^T
    mm_phase<KB1, K1, H4>(X, W1, G, H4 / 16, gwave, lane);
    gbar(bar);

    // ---- Phase C (block 0): LSTM1 cell + attention window
    if (blockIdx.x == 0) {
      lstm_ew(G, B1, C1, X + OFF_H1, tid, NTHR);
      __syncthreads();
      // p = exp(h1 @ Watt^T + batt)
      for (int i = tid; i < BB * 30; i += NTHR) {
        int b = i / 30, j = i - b * 30;
        float s = batt[j];
        const unsigned short* h1 = X + (size_t)b * LDX + OFF_H1;
        for (int k = 0; k < HID; ++k) s += bf2f(h1[k]) * Watt[j * HID + k];
        P[i] = expf(s);
      }
      __syncthreads();
      // kappa += dkappa
      for (int i = tid; i < BB * KATT; i += NTHR) {
        int b = i / KATT, k = i - b * KATT;
        KAP[i] += P[b * 30 + 20 + k];
      }
      __syncthreads();
      // phi[b,u] = sum_k alpha * exp(-beta * (kappa-u)^2), masked
      for (int i = tid; i < BB * TCC; i += NTHR) {
        int b = i / TCC, u = i - b * TCC;
        float ph = 0.f;
        for (int k = 0; k < KATT; ++k) {
          float d = KAP[b * KATT + k] - (float)u;
          ph += P[b * 30 + k] * expf(-P[b * 30 + 10 + k] * d * d);
        }
        PHI[i] = (u < clen[b]) ? ph : 0.f;
      }
      __syncthreads();
      // window[b,a] = sum_u phi[b,u] * (char_seq[b,u]==a)
      for (int i = tid; i < BB * 64; i += NTHR) {
        int b = i / 64, a = i - b * 64;
        float w = 0.f;
        if (a < ALPH)
          for (int u = 0; u < TCC; ++u)
            w += (cseq[b * TCC + u] == a) ? PHI[b * TCC + u] : 0.f;
        X[(size_t)b * LDX + OFF_WIN + a] = f2bf(w);
      }
    }
    gbar(bar);

    // ---- Phase D: LSTM2 gates
    mm_phase<KB2, K2, H4>(X, W2, G, H4 / 16, gwave, lane);
    gbar(bar);

    // ---- Phase E: LSTM2 cell (grid-wide)
    lstm_ew(G, B2, C2, X + OFF_H2, gtid, NBLK * NTHR);
    gbar(bar);

    // ---- Phase F: LSTM3 gates
    mm_phase<KB3, K3, H4>(X, W3, G, H4 / 16, gwave, lane);
    gbar(bar);

    // ---- Phase G: LSTM3 cell (grid-wide)
    lstm_ew(G, B3, C3, X + OFF_H3, gtid, NBLK * NTHR);
    gbar(bar);

    // ---- Phase H: GMM head   Z = X @ WG^T  (121 rows, padded to 128)
    mm_phase<KB3, K3, 128>(X, WG, Z, 128 / 16, gwave, lane);
    gbar(bar);
  }

  if (blockIdx.x == 0) out_ew(TT - 1, Z, BG, out, tid, NTHR);
}

// ---------------------------------------------------------------- launch
extern "C" void kernel_launch(void* const* d_in, const int* in_sizes, int n_in,
                              void* d_out, int out_size, void* d_ws, size_t ws_size,
                              hipStream_t stream) {
  (void)in_sizes; (void)n_in; (void)out_size; (void)ws_size;
  const float* xin  = (const float*)d_in[0];
  const int*   cseq = (const int*)d_in[1];
  const int*   clen = (const int*)d_in[2];
  const float* Wih1 = (const float*)d_in[3];
  const float* bih1 = (const float*)d_in[4];
  const float* Whh1 = (const float*)d_in[5];
  const float* bhh1 = (const float*)d_in[6];
  const float* Wih2 = (const float*)d_in[7];
  const float* bih2 = (const float*)d_in[8];
  const float* Whh2 = (const float*)d_in[9];
  const float* bhh2 = (const float*)d_in[10];
  const float* Wih3 = (const float*)d_in[11];
  const float* bih3 = (const float*)d_in[12];
  const float* Whh3 = (const float*)d_in[13];
  const float* bhh3 = (const float*)d_in[14];
  const float* Watt = (const float*)d_in[15];
  const float* batt = (const float*)d_in[16];
  const float* Wgmm = (const float*)d_in[17];
  const float* bgmm = (const float*)d_in[18];
  unsigned char* ws = (unsigned char*)d_ws;
  float* out = (float*)d_out;

  k_zero<<<256, 256, 0, stream>>>((unsigned*)ws);
  k_bias<<<(H4 + 255) / 256, 256, 0, stream>>>(bih1, bhh1, bih2, bhh2, bih3, bhh3,
                                               bgmm, ws);
  k_w1<<<(H4 * K1 + 255) / 256, 256, 0, stream>>>(Wih1, Whh1, ws);
  k_w2<<<(H4 * K2 + 255) / 256, 256, 0, stream>>>(Wih2, Whh2, ws);
  k_w3<<<(H4 * K3 + 255) / 256, 256, 0, stream>>>(Wih3, Whh3, ws);
  k_wg<<<(128 * K3 + 255) / 256, 256, 0, stream>>>(Wgmm, ws);
  k_rnn<<<NBLK, NTHR, 0, stream>>>(xin, cseq, clen, Watt, batt, ws, out);
}
